// ClusterNet_Contrast_75642964017169
// MI455X (gfx1250) — matrix operand order repaired and checked by reading, not compiled, not run on hardware
//
#include <hip/hip_runtime.h>

typedef __attribute__((ext_vector_type(16))) _Float16 v16h;
typedef __attribute__((ext_vector_type(8)))  _Float16 v8h;
typedef __attribute__((ext_vector_type(8)))  float    v8f;
typedef __attribute__((ext_vector_type(4)))  float    v4f;

// ---- problem constants (from reference setup_inputs) ----
constexpr int  BB   = 32;
constexpr int  HIN  = 256;
constexpr int  NTOK = 4096;   // 64*64 latent tokens
constexpr float EPS_SINK = 0.05f;

// ---- d_out layout (concatenated outputs, all float) ----
constexpr long long OFF_XBAR = 0;
constexpr long long OFF_Z    = (long long)BB * 3 * HIN * HIN;        //  6,291,456
constexpr long long OFF_LOG  = OFF_Z   + (long long)BB * 64 * NTOK;  // +8,388,608
constexpr long long OFF_IDX  = OFF_LOG + (long long)BB * 64 * NTOK;  // +8,388,608
constexpr long long OFF_PRO  = OFF_IDX + (long long)BB * NTOK;       // +  131,072

// =====================================================================
// conv1: 3->64, stride 2, pad 1, relu.  out: [32,64,128,128]
// =====================================================================
__global__ __launch_bounds__(256) void k_conv1(const float* __restrict__ x,
                                               const float* __restrict__ W1,
                                               const float* __restrict__ b1,
                                               float* __restrict__ h1) {
  __shared__ float wt[64 * 3 * 9];
  __shared__ float bs[64];
  int tid = threadIdx.x;
  for (int i = tid; i < 64 * 27; i += 256) wt[i] = W1[i];
  if (tid < 64) bs[tid] = b1[tid];
  __syncthreads();
  long long gid = (long long)blockIdx.x * 256 + tid;
  int xx = (int)(gid & 127), yy = (int)((gid >> 7) & 127);
  int co = (int)((gid >> 14) & 63), b = (int)(gid >> 20);
  float acc = bs[co];
  for (int ci = 0; ci < 3; ci++) {
    const float* xin = x + ((long long)(b * 3 + ci)) * HIN * HIN;
    const float* w = wt + (co * 3 + ci) * 9;
    for (int ky = 0; ky < 3; ky++) {
      int iy = 2 * yy - 1 + ky;
      if ((unsigned)iy >= 256u) continue;
      for (int kx = 0; kx < 3; kx++) {
        int ix = 2 * xx - 1 + kx;
        if ((unsigned)ix >= 256u) continue;
        acc += xin[iy * 256 + ix] * w[ky * 3 + kx];
      }
    }
  }
  h1[gid] = fmaxf(acc, 0.f);
}

// =====================================================================
// conv2: 64->64, stride 2, pad 1 (no relu).  h1[32,64,128,128] -> z[32,64,64,64]
// =====================================================================
__global__ __launch_bounds__(256) void k_conv2(const float* __restrict__ h1,
                                               const float* __restrict__ W2,
                                               const float* __restrict__ b2,
                                               float* __restrict__ z) {
  __shared__ float tin[32 * 17 * 17];
  int tid = threadIdx.x;
  int tile = blockIdx.x & 63, b = blockIdx.x >> 6;
  int TX0 = (tile & 7) * 8, TY0 = (tile >> 3) * 8;
  int p = tid & 63, px = p & 7, py = p >> 3, cob = (tid >> 6) * 16;
  float acc[16];
  for (int j = 0; j < 16; j++) acc[j] = 0.f;
  for (int ch = 0; ch < 2; ch++) {
    int ci0 = ch * 32;
    for (int e = tid; e < 32 * 289; e += 256) {
      int ci = e / 289, r = e % 289, iy = r / 17, ix = r % 17;
      int gy = TY0 * 2 - 1 + iy, gx = TX0 * 2 - 1 + ix;
      float v = 0.f;
      if ((unsigned)gy < 128u && (unsigned)gx < 128u)
        v = h1[(((long long)b * 64 + ci0 + ci) * 128 + gy) * 128 + gx];
      tin[e] = v;
    }
    __syncthreads();
    for (int j = 0; j < 16; j++) {
      int co = cob + j;
      const float* w = W2 + ((long long)co * 64 + ci0) * 9;
      float a = acc[j];
      for (int ci = 0; ci < 32; ci++) {
        const float* ld = tin + ci * 289;
        const float* wc = w + ci * 9;
        for (int ky = 0; ky < 3; ky++)
          for (int kx = 0; kx < 3; kx++)
            a += ld[(2 * py + ky) * 17 + 2 * px + kx] * __ldg(&wc[ky * 3 + kx]);
      }
      acc[j] = a;
    }
    __syncthreads();
  }
  for (int j = 0; j < 16; j++) {
    int co = cob + j;
    z[(((long long)b * 64 + co) * 64 + TY0 + py) * 64 + TX0 + px] =
        acc[j] + __ldg(&b2[co]);
  }
}

// =====================================================================
// conv3: up2(z) -> 64->64, stride 1, pad 1, relu.  out d: [32,64,128,128]
// =====================================================================
__global__ __launch_bounds__(256) void k_conv3(const float* __restrict__ z,
                                               const float* __restrict__ W3,
                                               const float* __restrict__ b3,
                                               float* __restrict__ dbuf) {
  __shared__ float zt[64 * 36];
  int tid = threadIdx.x;
  int tile = blockIdx.x & 255, b = blockIdx.x >> 8;
  int X0 = (tile & 15) * 8, Y0 = (tile >> 4) * 8;
  int zy0 = (Y0 == 0) ? 0 : ((Y0 - 1) >> 1);
  int zx0 = (X0 == 0) ? 0 : ((X0 - 1) >> 1);
  for (int e = tid; e < 64 * 36; e += 256) {
    int ci = e / 36, r = e % 36, ry = r / 6, rx = r % 6;
    int zy = zy0 + ry, zx = zx0 + rx;
    float v = 0.f;
    if (zy < 64 && zx < 64) v = z[(((long long)b * 64 + ci) * 64 + zy) * 64 + zx];
    zt[e] = v;
  }
  __syncthreads();
  int p = tid & 63, px = p & 7, py = p >> 3, cob = (tid >> 6) * 16;
  for (int j = 0; j < 16; j++) {
    int co = cob + j;
    const float* w = W3 + (long long)co * 576;
    float a = __ldg(&b3[co]);
    for (int ci = 0; ci < 64; ci++) {
      const float* lz = zt + ci * 36;
      const float* wc = w + ci * 9;
      for (int ky = 0; ky < 3; ky++) {
        int vy = Y0 + py + ky - 1;
        if ((unsigned)vy >= 128u) continue;
        int ly = (vy >> 1) - zy0;
        for (int kx = 0; kx < 3; kx++) {
          int vx = X0 + px + kx - 1;
          if ((unsigned)vx >= 128u) continue;
          a += lz[ly * 6 + (vx >> 1) - zx0] * __ldg(&wc[ky * 3 + kx]);
        }
      }
    }
    dbuf[(((long long)b * 64 + co) * 128 + Y0 + py) * 128 + X0 + px] = fmaxf(a, 0.f);
  }
}

// =====================================================================
// conv4: up2(d) -> 64->3, stride 1, pad 1.  out x_bar: [32,3,256,256]
// =====================================================================
__global__ __launch_bounds__(256) void k_conv4(const float* __restrict__ dbuf,
                                               const float* __restrict__ W4,
                                               const float* __restrict__ b4,
                                               float* __restrict__ xbar) {
  __shared__ float dt[64 * 100];
  __shared__ float wt[3 * 64 * 9];
  int tid = threadIdx.x;
  int tile = blockIdx.x & 255, b = blockIdx.x >> 8;
  int X0 = (tile & 15) * 16, Y0 = (tile >> 4) * 16;
  int dy0 = (Y0 == 0) ? 0 : ((Y0 - 1) >> 1);
  int dx0 = (X0 == 0) ? 0 : ((X0 - 1) >> 1);
  for (int i = tid; i < 1728; i += 256) wt[i] = W4[i];
  for (int e = tid; e < 6400; e += 256) {
    int ci = e / 100, r = e % 100, ry = r / 10, rx = r % 10;
    int dy = dy0 + ry, dx = dx0 + rx;
    float v = 0.f;
    if (dy < 128 && dx < 128)
      v = dbuf[(((long long)b * 64 + ci) * 128 + dy) * 128 + dx];
    dt[e] = v;
  }
  __syncthreads();
  int px = tid & 15, py = tid >> 4;
  float a0 = __ldg(&b4[0]), a1 = __ldg(&b4[1]), a2 = __ldg(&b4[2]);
  for (int ci = 0; ci < 64; ci++) {
    const float* ld = dt + ci * 100;
    for (int ky = 0; ky < 3; ky++) {
      int vy = Y0 + py + ky - 1;
      if ((unsigned)vy >= 256u) continue;
      int ly = (vy >> 1) - dy0;
      for (int kx = 0; kx < 3; kx++) {
        int vx = X0 + px + kx - 1;
        if ((unsigned)vx >= 256u) continue;
        float in = ld[ly * 10 + (vx >> 1) - dx0];
        int wi = ci * 9 + ky * 3 + kx;
        a0 += in * wt[wi];
        a1 += in * wt[576 + wi];
        a2 += in * wt[1152 + wi];
      }
    }
  }
  long long base = (((long long)b * 3) * 256 + Y0 + py) * 256 + X0 + px;
  xbar[base]          = a0;
  xbar[base + 65536]  = a1;
  xbar[base + 131072] = a2;
}

// =====================================================================
// Clustering: one workgroup (128 threads / 4 waves) per batch item.
// A-tiles are staged PRE-SWIZZLED into a 16x32 f16 A-fragment order (contiguous k = 16*half + j),
// so each lane's fragment load is two contiguous ds_load_b128 (no per-
// element gather stalls). B fragments are aligned 32B v16h loads.
// =====================================================================
__global__ void __launch_bounds__(128)
k_cluster(const float* __restrict__ zG,     // [B][64][4096]  (d_out z region)
          const float* __restrict__ pinit,  // [B][64][64]
          _Float16* __restrict__ znT,       // ws: [B][64][4096] f16, d-major
          float* __restrict__ qb,           // ws: [B][4096][64]
          float* __restrict__ scaleG,       // ws: [B][4096]
          float* __restrict__ logitsO,      // d_out logits [B][64][4096]
          float* __restrict__ idxO,         // d_out idx    [B][4096] (float)
          float* __restrict__ proO) {       // d_out protos [B][64][64]
  __shared__ float    protosF[64 * 64];                 // 16 KB
  __shared__ __align__(32) _Float16 protosH[64 * 64];   //  8 KB
  __shared__ __align__(32) unsigned char pool[24576];   // 24 KB, overlaid
  __shared__ float redBuf[128];
  __shared__ float rowsum[64];
  __shared__ unsigned char idxN[4096];

  _Float16* aT   = (_Float16*)pool;           // GEMM: 4 waves * 1024 halves (swizzled)
  float*    sT   = (float*)(pool + 8192);     // GEMM: 4 waves * 1024 floats
  float*    newp = (float*)pool;              // update: 64*64 (overlaid)
  float*    cnts = (float*)(pool + 16384);    // update: 64   (overlaid)

  int tid = threadIdx.x, b = blockIdx.x;
  int lane = tid & 31, wv = tid >> 5;
  const float* zb = zG + (long long)b * 262144;
  _Float16* znb = znT + (long long)b * 262144;
  float* qB  = qb + (long long)b * 262144;
  float* scb = scaleG + (long long)b * 4096;

  // phase 0a: token inverse norms
  for (int n = tid; n < 4096; n += 128) {
    float ss = 0.f;
    for (int c = 0; c < 64; c++) { float v = zb[c * 4096 + n]; ss += v * v; }
    scb[n] = 1.0f / fmaxf(sqrtf(ss), 1e-12f);
  }
  __syncthreads();
  // phase 0b: normalized tokens in f16 (vectorized: 2x v4f in, 1x v8h out)
  for (int e = tid * 8; e < 262144; e += 1024) {
    int n = e & 4095;
    v4f z0 = *(const v4f*)(zb + e);
    v4f z1 = *(const v4f*)(zb + e + 4);
    v4f s0 = *(const v4f*)(scb + n);
    v4f s1 = *(const v4f*)(scb + n + 4);
    v8h o;
    for (int u = 0; u < 4; u++) {
      o[u]     = (_Float16)(z0[u] * s0[u]);
      o[u + 4] = (_Float16)(z1[u] * s1[u]);
    }
    *(v8h*)(znb + e) = o;
  }
  // phase 0c: l2-normalized initial prototypes
  if (tid < 64) {
    const float* p0 = pinit + ((long long)b * 64 + tid) * 64;
    float ss = 0.f;
    for (int d = 0; d < 64; d++) { float v = p0[d]; ss += v * v; }
    float sc = 1.0f / fmaxf(sqrtf(ss), 1e-12f);
    for (int d = 0; d < 64; d++) {
      float v = p0[d] * sc;
      protosF[tid * 64 + d] = v;
      protosH[tid * 64 + d] = (_Float16)v;
    }
  }
  __syncthreads();

  for (int it = 0; it < 3; it++) {
    // ---------- scores (WMMA) + softmax + exp(p/eps) -> Q ----------
    // B fragments: 4 cluster tiles x 2 K-chunks, aligned v16h loads.
    v16h bf[4][2];
    {
      int cl = lane & 15;
      int kb0 = (lane < 16) ? 0 : 16;
      for (int nt = 0; nt < 4; nt++)
        for (int ch = 0; ch < 2; ch++)
          bf[nt][ch] = *(const v16h*)(protosH + (nt * 16 + cl) * 64 + ch * 32 + kb0);
    }
    _Float16* aw = aT + wv * 1024;
    float*    sw = sT + wv * 1024;
    for (int t = wv; t < 256; t += 4) {           // 256 token tiles of 16
      // stage A tile pre-swizzled into fragment order:
      // dest element for (kd=d, m): chunk=d>>5; k32=d&31;
      //   lane-half = k32>>4; j = k32&15
      for (int d = lane; d < 64; d += 32) {
        v16h row = *(const v16h*)(znb + d * 4096 + t * 16);  // 16 tokens of dim d
        int ch = d >> 5, k32 = d & 31;
        int lh = k32 >> 4;
        int j0 = k32 & 15;
        _Float16* dst = aw + ch * 512 + lh * 256 + j0;
        for (int m = 0; m < 16; m++) dst[m * 16] = row[m];
      }
      __builtin_amdgcn_wave_barrier();
      v16h af0 = *(const v16h*)(aw + lane * 16);        // chunk 0 fragment
      v16h af1 = *(const v16h*)(aw + 512 + lane * 16);  // chunk 1 fragment
      for (int nt = 0; nt < 4; nt++) {
        v8f c = {0.f, 0.f, 0.f, 0.f, 0.f, 0.f, 0.f, 0.f};
        c = __builtin_amdgcn_wmma_f32_16x16x32_f16(false, af0, false, bf[nt][0],
                                                   (short)0, c, false, false);
        c = __builtin_amdgcn_wmma_f32_16x16x32_f16(false, af1, false, bf[nt][1],
                                                   (short)0, c, false, false);
        int cluster = nt * 16 + (lane & 15);
        int mb = (lane < 16) ? 0 : 8;
        for (int r = 0; r < 8; r++) sw[(mb + r) * 64 + cluster] = c[r];
      }
      __builtin_amdgcn_wave_barrier();
      if (lane < 16) {                            // softmax row + exp(p/eps)
        v4f* r4 = (v4f*)(sw + lane * 64);
        v4f buf[16];
        float mx = -3.4e38f;
        for (int i = 0; i < 16; i++) {
          buf[i] = r4[i];
          for (int u = 0; u < 4; u++) mx = fmaxf(mx, buf[i][u]);
        }
        float ssum = 0.f;
        for (int i = 0; i < 16; i++)
          for (int u = 0; u < 4; u++) {
            float e0 = __expf(buf[i][u] - mx);
            buf[i][u] = e0; ssum += e0;
          }
        float inv = (1.0f / EPS_SINK) / ssum;
        for (int i = 0; i < 16; i++) {
          for (int u = 0; u < 4; u++) buf[i][u] = __expf(buf[i][u] * inv);
          r4[i] = buf[i];
        }
      }
      __builtin_amdgcn_wave_barrier();
      for (int e = lane; e < 256; e += 32)        // s-tile -> Q, 16B stores
        *(v4f*)(qB + t * 1024 + e * 4) = *(const v4f*)(sw + e * 4);
      __builtin_amdgcn_wave_barrier();
    }
    __syncthreads();

    // ---------- sinkhorn (global norm cancels into first row-norm) ----------
    for (int si = 0; si < 3; si++) {
      {                                           // rowsum[k] = sum_n Q[n][k]
        int k = tid & 63, g = tid >> 6;
        float loc = 0.f;
        for (int n = g; n < 4096; n += 2) loc += qB[n * 64 + k];
        redBuf[tid] = loc;
        __syncthreads();
        if (tid < 64) rowsum[tid] = redBuf[tid] + redBuf[tid + 64];
        __syncthreads();
      }
      for (int n = tid; n < 4096; n += 128) {     // row-div then col-div per token
        v4f* row4 = (v4f*)(qB + n * 64);
        v4f tmp[16];
        float cs = 0.f;
        for (int i = 0; i < 16; i++) {
          v4f v = row4[i];
          for (int u = 0; u < 4; u++) {
            float w = v[u] / (rowsum[i * 4 + u] * 64.0f);
            v[u] = w; cs += w;
          }
          tmp[i] = v;
        }
        float inv = 1.0f / (cs * 4096.0f);
        for (int i = 0; i < 16; i++) {
          v4f v = tmp[i];
          for (int u = 0; u < 4; u++) v[u] *= inv;
          row4[i] = v;
        }
      }
      __syncthreads();
    }

    // ---------- q = Q*N, argmax; write outputs on last iter ----------
    for (int n = tid; n < 4096; n += 128) {
      const v4f* row4 = (const v4f*)(qB + n * 64);
      float best = -3.4e38f; int bi = 0;
      for (int i = 0; i < 16; i++) {
        v4f v = row4[i];
        for (int u = 0; u < 4; u++) {
          int k = i * 4 + u;
          float val = v[u] * 4096.0f;
          if (it == 2) logitsO[(long long)b * 262144 + k * 4096 + n] = val;
          if (val > best) { best = val; bi = k; }
        }
      }
      idxN[n] = (unsigned char)bi;
      if (it == 2) idxO[(long long)b * 4096 + n] = (float)bi;
    }
    __syncthreads();

    // ---------- prototype update: deterministic k-major scan ----------
    {
      int k = tid & 63, dh = tid >> 6;            // dh selects d-half
      float acc[32];
      for (int d = 0; d < 32; d++) acc[d] = 0.f;
      float cnt = 0.f;
      for (int n = 0; n < 4096; n++) {
        if (idxN[n] == (unsigned char)k) {
          cnt += 1.f;
          const float* zcol = zb + (dh * 32) * 4096 + n;
          for (int d = 0; d < 32; d++) acc[d] += zcol[d * 4096];
        }
      }
      if (dh == 0) cnts[k] = cnt;
      for (int d = 0; d < 32; d++) newp[k * 64 + dh * 32 + d] = acc[d];
    }
    __syncthreads();
    if (tid < 64) {
      int k = tid;
      float c = cnts[k], ss = 0.f, pr[64];
      for (int d = 0; d < 64; d++) {
        float v = (c > 0.f) ? newp[k * 64 + d] / c : 0.f;
        pr[d] = v; ss += v * v;
      }
      float sc = 1.0f / fmaxf(sqrtf(ss), 1e-12f);
      for (int d = 0; d < 64; d++) {
        float v = pr[d] * sc;
        protosF[k * 64 + d] = v;
        protosH[k * 64 + d] = (_Float16)v;
      }
    }
    __syncthreads();
  }
  for (int i = tid; i < 4096; i += 128) proO[(long long)b * 4096 + i] = protosF[i];
}

// =====================================================================
extern "C" void kernel_launch(void* const* d_in, const int* in_sizes, int n_in,
                              void* d_out, int out_size, void* d_ws, size_t ws_size,
                              hipStream_t stream) {
  (void)in_sizes; (void)n_in; (void)out_size; (void)ws_size;
  const float* x     = (const float*)d_in[0];
  const float* W1    = (const float*)d_in[1];
  const float* b1    = (const float*)d_in[2];
  const float* W2    = (const float*)d_in[3];
  const float* b2    = (const float*)d_in[4];
  const float* W3    = (const float*)d_in[5];
  const float* b3    = (const float*)d_in[6];
  const float* W4    = (const float*)d_in[7];
  const float* b4    = (const float*)d_in[8];
  const float* pinit = (const float*)d_in[9];

  float* out = (float*)d_out;
  float* xbar   = out + OFF_XBAR;
  float* z      = out + OFF_Z;
  float* logits = out + OFF_LOG;
  float* idxo   = out + OFF_IDX;
  float* proto  = out + OFF_PRO;

  // workspace layout (floats): h1/d (reused) | znT f16 | q | scale
  float* wsF = (float*)d_ws;
  float*    h1buf  = wsF;                                  // 33,554,432 f
  _Float16* znT    = (_Float16*)(wsF + 33554432);          //  8,388,608 h
  float*    qbuf   = wsF + 33554432 + 4194304;             //  8,388,608 f
  float*    scaleB = qbuf + 8388608;                       //    131,072 f

  k_conv1<<<131072, 256, 0, stream>>>(x, W1, b1, h1buf);
  k_conv2<<<2048,   256, 0, stream>>>(h1buf, W2, b2, z);
  k_conv3<<<8192,   256, 0, stream>>>(z, W3, b3, h1buf);   // d reuses h1 buffer
  k_conv4<<<8192,   256, 0, stream>>>(h1buf, W4, b4, xbar);
  k_cluster<<<32, 128, 0, stream>>>(z, pinit, znT, qbuf, scaleB, logits, idxo, proto);
}
